// MultiHeadAttention4AKT_16475494547875
// MI455X (gfx1250) — matrix operand order repaired and checked by reading, not compiled, not run on hardware
//
#include <hip/hip_runtime.h>
#include <hip/hip_bf16.h>

// ---------------------------------------------------------------------------
// MultiHeadAttention4AKT for MI455X (gfx1250, wave32, WMMA bf16,
// async global->LDS staging, double-buffered GEMM slabs)
// B=8, S=512, D=1024, H=16, DH=64
// ---------------------------------------------------------------------------

#define B_  8
#define S_  512
#define D_  1024
#define H_  16
#define DH_ 64
#define M_  (B_ * S_)            // 4096 rows for projection GEMMs

typedef __attribute__((ext_vector_type(16))) __bf16 v16bf;
typedef __attribute__((ext_vector_type(8)))  float  v8f;

struct U32x8 { uint4 a, b; };

static __device__ __forceinline__ v16bf frag_from(uint4 lo, uint4 hi) {
    U32x8 t; t.a = lo; t.b = hi;
    return __builtin_bit_cast(v16bf, t);
}

static __device__ __forceinline__ unsigned short f2bf(float f) {
    unsigned u = __builtin_bit_cast(unsigned, f);
    unsigned r = u + 0x7FFFu + ((u >> 16) & 1u);   // round-to-nearest-even
    return (unsigned short)(r >> 16);
}

// ---- CDNA5 async global->LDS copy (ASYNCcnt path, ISA 15.18.3 op 98) ------
static __device__ __forceinline__ void async_load_b128(unsigned lds_byte_addr,
                                                       const void* gaddr) {
    asm volatile("global_load_async_to_lds_b128 %0, %1, off"
                 :
                 : "v"(lds_byte_addr),
                   "v"((unsigned long long)(uintptr_t)gaddr)
                 : "memory");
}

static __device__ __forceinline__ void wait_async0() {
#if defined(__has_builtin)
#if __has_builtin(__builtin_amdgcn_s_wait_asynccnt)
    __builtin_amdgcn_s_wait_asynccnt(0);
    return;
#endif
#endif
    asm volatile("s_wait_asynccnt 0x0" ::: "memory");
}

static __device__ __forceinline__ unsigned lds_addr_of(const void* p) {
    return (unsigned)(uintptr_t)p;   // low 32 bits of LDS aperture = offset
}

// --------------------------- f32 -> bf16 convert ---------------------------
__global__ __launch_bounds__(256) void cvt_f32_bf16(
    const float* __restrict__ src, unsigned short* __restrict__ dst, int n) {
    int i = (blockIdx.x * 256 + threadIdx.x) * 4;
    if (i < n) {
        float4 v = *(const float4*)(src + i);
        unsigned p0 = (unsigned)f2bf(v.x) | ((unsigned)f2bf(v.y) << 16);
        unsigned p1 = (unsigned)f2bf(v.z) | ((unsigned)f2bf(v.w) << 16);
        *(uint2*)(dst + i) = make_uint2(p0, p1);
    }
}

// ----------------------- projection GEMM (Y = X W^T + b) -------------------
// grid (M/128, D/64, 3)  block 256 (8 waves), 2x2 16x16 WMMA tiles per wave.
// z=0: q->qh[bh][s][64], z=1: k->kh[bh][s][64], z=2: v->vt[bh][64][s]
#define A_ELEMS (128 * 32)
#define B_ELEMS (64 * 32)

__global__ __launch_bounds__(256) void proj_gemm(
    const unsigned short* __restrict__ qb, const unsigned short* __restrict__ kb,
    const unsigned short* __restrict__ vb,
    const unsigned short* __restrict__ Wkb, const unsigned short* __restrict__ Wvb,
    const float* __restrict__ bk, const float* __restrict__ bv,
    unsigned short* __restrict__ qh, unsigned short* __restrict__ kh,
    unsigned short* __restrict__ vt) {

    const int z = blockIdx.z;
    const unsigned short* X = (z == 0) ? qb : (z == 1) ? kb : vb;
    const unsigned short* W = (z == 2) ? Wvb : Wkb;
    const float* bias = (z == 2) ? bv : bk;

    __shared__ unsigned short sA[2 * A_ELEMS];
    __shared__ unsigned short sB[2 * B_ELEMS];

    const int tid  = threadIdx.x;
    const int lane = tid & 31;
    const int wave = tid >> 5;
    const int wm = wave >> 1, wn = wave & 1;
    const int g = lane >> 4, ln = lane & 15;
    const int bm0 = blockIdx.x * 128;
    const int bn0 = blockIdx.y * 64;

    // staging geometry (per thread, constant)
    const int rowA = tid >> 2, qA = tid & 3;           // + second chunk row+64
    const int eA0 = rowA * 32 + qA * 8;
    const int eA1 = eA0 + 64 * 32;
    const unsigned aBase = lds_addr_of(sA);
    const unsigned bBase = lds_addr_of(sB);
    const unsigned short* gA0 = X + (size_t)(bm0 + rowA) * D_ + qA * 8;
    const unsigned short* gA1 = gA0 + (size_t)64 * D_;
    const unsigned short* gB  = W + (size_t)(bn0 + rowA) * D_ + qA * 8;

    v8f acc[2][2] = {};

    // prime buffer 0
    async_load_b128(aBase + eA0 * 2, gA0);
    async_load_b128(aBase + eA1 * 2, gA1);
    async_load_b128(bBase + eA0 * 2, gB);

    for (int it = 0; it < D_ / 32; ++it) {
        const int cur = it & 1;
        wait_async0();
        __syncthreads();
        if (it + 1 < D_ / 32) {                    // prefetch next slab (async)
            const int nxt = cur ^ 1;
            const int k1 = (it + 1) * 32;
            async_load_b128(aBase + (nxt * A_ELEMS + eA0) * 2, gA0 + k1);
            async_load_b128(aBase + (nxt * A_ELEMS + eA1) * 2, gA1 + k1);
            async_load_b128(bBase + (nxt * B_ELEMS + eA0) * 2, gB + k1);
        }

        const unsigned short* cA = sA + cur * A_ELEMS;
        const unsigned short* cB = sB + cur * B_ELEMS;
        v16bf a[2], b[2];
#pragma unroll
        for (int i = 0; i < 2; ++i) {
            const unsigned short* p = cA + (wm * 32 + i * 16 + ln) * 32;
            a[i] = frag_from(*(const uint4*)(p + g * 8),
                             *(const uint4*)(p + 16 + g * 8));
        }
#pragma unroll
        for (int j = 0; j < 2; ++j) {
            const unsigned short* p = cB + (wn * 32 + j * 16 + ln) * 32;
            b[j] = frag_from(*(const uint4*)(p + g * 8), *(const uint4*)(p + 16 + g * 8));
        }
#pragma unroll
        for (int i = 0; i < 2; ++i)
#pragma unroll
            for (int j = 0; j < 2; ++j)
                acc[i][j] = __builtin_amdgcn_wmma_f32_16x16x32_bf16(
                    false, a[i], false, b[j], (short)0, acc[i][j], false, false);
    }

#pragma unroll
    for (int i = 0; i < 2; ++i)
#pragma unroll
        for (int j = 0; j < 2; ++j)
#pragma unroll
            for (int r = 0; r < 8; ++r) {
                int gm = bm0 + wm * 32 + i * 16 + r + 8 * g;
                int gn = bn0 + wn * 32 + j * 16 + ln;
                unsigned short o = f2bf(acc[i][j][r] + bias[gn]);
                int bb = gm >> 9, s = gm & 511;
                int hh = gn >> 6, d = gn & 63;
                size_t bh = (size_t)(bb * H_ + hh);
                if (z == 0)      qh[(bh * S_ + s) * DH_ + d] = o;
                else if (z == 1) kh[(bh * S_ + s) * DH_ + d] = o;
                else             vt[(bh * DH_ + d) * S_ + s] = o;
            }
}

// ----------------------- output GEMM (out = X Wo^T + bo, f32) --------------
__global__ __launch_bounds__(256) void out_gemm(
    const unsigned short* __restrict__ X, const unsigned short* __restrict__ W,
    const float* __restrict__ bias, float* __restrict__ out) {

    __shared__ unsigned short sA[2 * A_ELEMS];
    __shared__ unsigned short sB[2 * B_ELEMS];

    const int tid  = threadIdx.x;
    const int lane = tid & 31;
    const int wave = tid >> 5;
    const int wm = wave >> 1, wn = wave & 1;
    const int g = lane >> 4, ln = lane & 15;
    const int bm0 = blockIdx.x * 128;
    const int bn0 = blockIdx.y * 64;

    const int rowA = tid >> 2, qA = tid & 3;
    const int eA0 = rowA * 32 + qA * 8;
    const int eA1 = eA0 + 64 * 32;
    const unsigned aBase = lds_addr_of(sA);
    const unsigned bBase = lds_addr_of(sB);
    const unsigned short* gA0 = X + (size_t)(bm0 + rowA) * D_ + qA * 8;
    const unsigned short* gA1 = gA0 + (size_t)64 * D_;
    const unsigned short* gB  = W + (size_t)(bn0 + rowA) * D_ + qA * 8;

    v8f acc[2][2] = {};

    async_load_b128(aBase + eA0 * 2, gA0);
    async_load_b128(aBase + eA1 * 2, gA1);
    async_load_b128(bBase + eA0 * 2, gB);

    for (int it = 0; it < D_ / 32; ++it) {
        const int cur = it & 1;
        wait_async0();
        __syncthreads();
        if (it + 1 < D_ / 32) {
            const int nxt = cur ^ 1;
            const int k1 = (it + 1) * 32;
            async_load_b128(aBase + (nxt * A_ELEMS + eA0) * 2, gA0 + k1);
            async_load_b128(aBase + (nxt * A_ELEMS + eA1) * 2, gA1 + k1);
            async_load_b128(bBase + (nxt * B_ELEMS + eA0) * 2, gB + k1);
        }

        const unsigned short* cA = sA + cur * A_ELEMS;
        const unsigned short* cB = sB + cur * B_ELEMS;
        v16bf a[2], b[2];
#pragma unroll
        for (int i = 0; i < 2; ++i) {
            const unsigned short* p = cA + (wm * 32 + i * 16 + ln) * 32;
            a[i] = frag_from(*(const uint4*)(p + g * 8),
                             *(const uint4*)(p + 16 + g * 8));
        }
#pragma unroll
        for (int j = 0; j < 2; ++j) {
            const unsigned short* p = cB + (wn * 32 + j * 16 + ln) * 32;
            b[j] = frag_from(*(const uint4*)(p + g * 8), *(const uint4*)(p + 16 + g * 8));
        }
#pragma unroll
        for (int i = 0; i < 2; ++i)
#pragma unroll
            for (int j = 0; j < 2; ++j)
                acc[i][j] = __builtin_amdgcn_wmma_f32_16x16x32_bf16(
                    false, a[i], false, b[j], (short)0, acc[i][j], false, false);
    }

#pragma unroll
    for (int i = 0; i < 2; ++i)
#pragma unroll
        for (int j = 0; j < 2; ++j)
#pragma unroll
            for (int r = 0; r < 8; ++r) {
                int gm = bm0 + wm * 32 + i * 16 + r + 8 * g;
                int gn = bn0 + wn * 32 + j * 16 + ln;
                out[(size_t)gm * D_ + gn] = acc[i][j][r] + bias[gn];
            }
}

// ------------------------------- attention ---------------------------------
// grid (S/16, B*H), block 64 (2 waves). One 16-query-row block per WG.
#define SSTR 520   // padded f32/bf16 row stride (bank rotation, 16B aligned)

__global__ __launch_bounds__(64) void attn_kernel(
    const unsigned short* __restrict__ qh, const unsigned short* __restrict__ kh,
    const unsigned short* __restrict__ vt, const float* __restrict__ pdiff,
    const float* __restrict__ gammas, unsigned short* __restrict__ concat) {

    __shared__ float          sS[16 * SSTR];   // scaled scores (f32)
    __shared__ unsigned short sP[16 * SSTR];   // final probs (bf16)

    const int tid  = threadIdx.x;
    const int lane = tid & 31;
    const int wave = tid >> 5;
    const int g = lane >> 4, ln = lane & 15;
    const int qt = blockIdx.x;          // query tile 0..31
    const int q0 = qt * 16;
    const int bh = blockIdx.y;          // 0..127
    const int b  = bh >> 4, h = bh & 15;

    {   // zero probs buffer (upper triangle / rounding slack must be 0)
        unsigned* pz = (unsigned*)sP;
        for (int i = tid; i < 16 * SSTR / 2; i += 64) pz[i] = 0u;
    }

    // prefetch pdiff rows needed in phase 2 (global_prefetch)
    for (int m = wave; m < 16; m += 2)
        __builtin_prefetch(pdiff + ((size_t)b * S_ + q0 + m) * S_ + lane * 16, 0, 1);

    // ---- phase 1: scores = (Q Kt)/8 into LDS, causal j-tiles only ----
    const unsigned short* qrow  = qh + ((size_t)bh * S_ + q0) * DH_;
    const unsigned short* kbase = kh + (size_t)bh * S_ * DH_;

    v16bf afr[2];
#pragma unroll
    for (int kk = 0; kk < 2; ++kk) {
        const unsigned short* p = qrow + (size_t)ln * DH_ + kk * 32;
        afr[kk] = frag_from(*(const uint4*)(p + g * 8),
                            *(const uint4*)(p + 16 + g * 8));
    }
    for (int jt = wave; jt <= qt; jt += 2) {
        int j0 = jt * 16;
        v8f acc = {};
#pragma unroll
        for (int kk = 0; kk < 2; ++kk) {
            const unsigned short* p =
                kbase + (size_t)(j0 + ln) * DH_ + kk * 32;
            v16bf bf = frag_from(*(const uint4*)(p + g * 8), *(const uint4*)(p + 16 + g * 8));
            acc = __builtin_amdgcn_wmma_f32_16x16x32_bf16(
                false, afr[kk], false, bf, (short)0, acc, false, false);
        }
#pragma unroll
        for (int r = 0; r < 8; ++r)
            sS[(r + 8 * g) * SSTR + j0 + ln] = acc[r] * 0.125f;
    }
    __syncthreads();

    // ---- phase 2: per-row softmax / cumsum / decay / softmax ----
    float gam = gammas[h];
    float gamma = -((gam > 20.f) ? gam : log1pf(expf(gam)));  // -softplus

    for (int m = wave; m < 16; m += 2) {      // one full wave per row
        int qi = q0 + m;
        int jbase = lane * 16;
        const float* srow = sS + m * SSTR + jbase;
        const float* prow = pdiff + ((size_t)b * S_ + qi) * S_ + jbase;
        float s[16], pd[16];
#pragma unroll
        for (int i4 = 0; i4 < 4; ++i4) {
            float4 v = *(const float4*)(srow + i4 * 4);
            s[i4*4+0]=v.x; s[i4*4+1]=v.y; s[i4*4+2]=v.z; s[i4*4+3]=v.w;
            float4 w = *(const float4*)(prow + i4 * 4);
            pd[i4*4+0]=w.x; pd[i4*4+1]=w.y; pd[i4*4+2]=w.z; pd[i4*4+3]=w.w;
        }
        // softmax #1 (causal mask)
        float mx = -3.0e38f;
#pragma unroll
        for (int i = 0; i < 16; ++i) {
            float sv = ((jbase + i) <= qi) ? s[i] : -1.0e32f;
            s[i] = sv;
            mx = fmaxf(mx, sv);
        }
        for (int o = 16; o > 0; o >>= 1) mx = fmaxf(mx, __shfl_xor(mx, o, 32));
        float p[16], lsum = 0.f;
#pragma unroll
        for (int i = 0; i < 16; ++i) { p[i] = expf(s[i] - mx); lsum += p[i]; }
        float tsum = lsum;
        for (int o = 16; o > 0; o >>= 1) tsum += __shfl_xor(tsum, o, 32);
        float inv = 1.f / tsum;
        // inclusive cumsum across 512 cols (lane-local + wave scan)
        float c[16], run = 0.f;
#pragma unroll
        for (int i = 0; i < 16; ++i) { run += p[i] * inv; c[i] = run; }
        float x = run;
        for (int d2 = 1; d2 < 32; d2 <<= 1) {
            float y = __shfl_up(x, d2, 32);
            if (lane >= d2) x += y;
        }
        float tot  = __shfl(x, 31, 32);
        float excl = x - run;
        // decay reweight + softmax #2
        float mx2 = -3.0e38f, s2[16];
#pragma unroll
        for (int i = 0; i < 16; ++i) {
            int j = jbase + i;
            float cum  = excl + c[i];
            float posv = fabsf((float)(qi - j));
            float dist = sqrtf(fmaxf((tot - cum) * posv, 0.f));
            float sg   = 1.f / (1.f + expf(-pd[i]));
            float eff  = expf(dist * gamma * expf(sg));
            eff = fminf(fmaxf(eff, 1e-5f), 1e5f);
            float v2 = (j <= qi) ? s[i] * eff : -1.0e32f;
            s2[i] = v2;
            mx2 = fmaxf(mx2, v2);
        }
        for (int o = 16; o > 0; o >>= 1) mx2 = fmaxf(mx2, __shfl_xor(mx2, o, 32));
        float l2 = 0.f;
#pragma unroll
        for (int i = 0; i < 16; ++i) { s2[i] = expf(s2[i] - mx2); l2 += s2[i]; }
        for (int o = 16; o > 0; o >>= 1) l2 += __shfl_xor(l2, o, 32);
        float inv2 = (qi == 0) ? 0.f : 1.f / l2;   // zero_pad first query row
        unsigned* dp = (unsigned*)(sP + m * SSTR + jbase);
#pragma unroll
        for (int i2 = 0; i2 < 8; ++i2)
            dp[i2] = (unsigned)f2bf(s2[2*i2] * inv2) |
                     ((unsigned)f2bf(s2[2*i2+1] * inv2) << 16);
    }
    __syncthreads();

    // ---- phase 3: out = P V  (wave w -> d columns [w*32, w*32+32)) ----
    const unsigned short* vbase = vt + (size_t)bh * DH_ * S_;
    const int jmax = (qt + 1) * 16;
    const int d0 = wave * 32;
    v8f o[2] = {};
    for (int k0 = 0; k0 < jmax; k0 += 32) {
        const unsigned short* ap = sP + (size_t)ln * SSTR + k0;
        v16bf af = frag_from(*(const uint4*)(ap + g * 8),
                             *(const uint4*)(ap + 16 + g * 8));
#pragma unroll
        for (int t = 0; t < 2; ++t) {
            const unsigned short* vp =
                vbase + (size_t)(d0 + t * 16 + ln) * S_ + k0;
            v16bf bf = frag_from(*(const uint4*)(vp + g * 8), *(const uint4*)(vp + 16 + g * 8));
            o[t] = __builtin_amdgcn_wmma_f32_16x16x32_bf16(
                false, af, false, bf, (short)0, o[t], false, false);
        }
    }
    unsigned short* cc = concat + ((size_t)b * S_ + q0) * D_ + h * DH_;
#pragma unroll
    for (int t = 0; t < 2; ++t)
#pragma unroll
        for (int r = 0; r < 8; ++r)
            cc[(size_t)(r + 8 * g) * D_ + d0 + t * 16 + ln] = f2bf(o[t][r]);
}

// ------------------------------- launcher ----------------------------------
extern "C" void kernel_launch(void* const* d_in, const int* in_sizes, int n_in,
                              void* d_out, int out_size, void* d_ws, size_t ws_size,
                              hipStream_t stream) {
    (void)in_sizes; (void)n_in; (void)out_size; (void)ws_size;

    const float* q      = (const float*)d_in[0];
    const float* k      = (const float*)d_in[1];
    const float* v      = (const float*)d_in[2];
    const float* pdiff  = (const float*)d_in[3];
    const float* Wk     = (const float*)d_in[4];
    const float* bk     = (const float*)d_in[5];
    const float* Wv     = (const float*)d_in[6];
    const float* bv     = (const float*)d_in[7];
    const float* Wo     = (const float*)d_in[8];
    const float* bo     = (const float*)d_in[9];
    const float* gammas = (const float*)d_in[10];
    float* out = (float*)d_out;

    // workspace layout
    constexpr size_t NQ = (size_t)M_ * D_;       // 4,194,304 elems
    constexpr size_t NW = (size_t)D_ * D_;       // 1,048,576 elems
    char* ws = (char*)d_ws;
    unsigned short* qb  = (unsigned short*)(ws);
    unsigned short* kb  = qb  + NQ;
    unsigned short* vb  = kb  + NQ;
    unsigned short* Wkb = vb  + NQ;
    unsigned short* Wvb = Wkb + NW;
    unsigned short* Wob = Wvb + NW;
    unsigned short* qhb = Wob + NW;
    unsigned short* khb = qhb + NQ;
    unsigned short* vtb = khb + NQ;
    unsigned short* cat = vtb + NQ;              // concat [B,S,D] bf16

    // 1) convert inputs & weights to bf16
    cvt_f32_bf16<<<dim3((unsigned)(NQ / 1024)), 256, 0, stream>>>(q,  qb,  (int)NQ);
    cvt_f32_bf16<<<dim3((unsigned)(NQ / 1024)), 256, 0, stream>>>(k,  kb,  (int)NQ);
    cvt_f32_bf16<<<dim3((unsigned)(NQ / 1024)), 256, 0, stream>>>(v,  vb,  (int)NQ);
    cvt_f32_bf16<<<dim3((unsigned)(NW / 1024)), 256, 0, stream>>>(Wk, Wkb, (int)NW);
    cvt_f32_bf16<<<dim3((unsigned)(NW / 1024)), 256, 0, stream>>>(Wv, Wvb, (int)NW);
    cvt_f32_bf16<<<dim3((unsigned)(NW / 1024)), 256, 0, stream>>>(Wo, Wob, (int)NW);

    // 2) fused q/k/v projections (z: 0=q, 1=k, 2=v-transposed)
    proj_gemm<<<dim3(M_ / 128, D_ / 64, 3), 256, 0, stream>>>(
        qb, kb, vb, Wkb, Wvb, bk, bv, qhb, khb, vtb);

    // 3) attention (per 16-query-row block per (b,h))
    attn_kernel<<<dim3(S_ / 16, B_ * H_), 64, 0, stream>>>(
        qhb, khb, vtb, pdiff, gammas, cat);

    // 4) output projection -> f32
    out_gemm<<<dim3(M_ / 128, D_ / 64), 256, 0, stream>>>(cat, Wob, bo, out);
}
